// ExpertNet_23407571763310
// MI455X (gfx1250) — compile-verified, matrix operand order checked by reading, not run on hardware
//
#include <hip/hip_runtime.h>

// ---------------------------------------------------------------------------
// ExpertNet fused kernel for MI455X (gfx1250), wave32 + WMMA bf16.
//   preds[n,c] = sum_k q[n,k] * ( relu(z@W1[k]+b1[k]) @ W2[k] + b2[k] )[c]
// Fusion: scale h rows by q[n,k] BEFORE GEMM2 so GEMM2 accumulates across
// kk2, hc AND k into one persistent f32 accumulator; add sum_k q*b2 at end.
// Data path: z tile resident in LDS (bf16), W1 double-buffered KT-slices,
// W2 staged one full 64x128 chunk per H-chunk, mu copied via
// global_load_async_to_lds_b128 (ASYNCcnt), weights prefetched ahead.
// Staging issues all global b128 loads before converting/storing so several
// loads are in flight (descending s_wait_loadcnt instead of wait-0 per load).
// ---------------------------------------------------------------------------

typedef __bf16 bf16_t;
typedef __attribute__((ext_vector_type(16))) __bf16 v16bf;
typedef __attribute__((ext_vector_type(8)))  __bf16 v8bf;
typedef __attribute__((ext_vector_type(4)))  __bf16 v4bf;
typedef __attribute__((ext_vector_type(8)))  float  v8f;
typedef __attribute__((ext_vector_type(4)))  float  v4f;

#define N_  32768
#define NZ_ 512
#define H_  1024
#define C_  128
#define K_  8

#define TM  128   // rows per block
#define KT  32    // WMMA K-slice (bf16 16x16x32)
#define HT  64    // H-chunk per GEMM1->GEMM2 round

#define LDZ (NZ_ + 8)  // 520 bf16 -> 1040 B row pitch (16B aligned)
#define LDW (KT + 8)   // 40  bf16 -> 80 B   (W1 slice, [h][d])
#define LDC (HT + 8)   // 72  bf16 -> 144 B  (W2 chunk [c][h] and h_s [m][h])

// LDS carving (dynamic shared memory) -- all offsets 16B aligned.
#define SM_Q     0                          // 128*8 f32          = 4096
#define SM_MU    4096                       // 8*512 f32          = 16384
#define SM_Z     (SM_MU + 16384)            // 128*520 bf16       = 133120
#define SM_W1    (SM_Z + TM * LDZ * 2)      // 2 x 64*40 bf16     = 10240
#define SM_W2    (SM_W1 + 2 * HT * LDW * 2) // 128*72 bf16        = 18432
#define SM_H     (SM_W2 + C_ * LDC * 2)     // 128*72 bf16        = 18432
#define SM_BYTES (SM_H + TM * LDC * 2)      // 200704 B < 320 KB/WGP

// reuse_a / reuse_b must be frontend integer constants -> literal macro arg.
#define WMMA_BF16(A, B, C, RA)                                          \
  __builtin_amdgcn_wmma_f32_16x16x32_bf16(false, (A), false, (B),       \
                                          (short)0, (C), (RA), false)

// Load one 16x32 bf16 operand fragment from LDS laid out [16 rows][K contig].
// CDNA5 16-bit A/B layout: lanes 0-15 hold K=0..7 (v0..3) and K=16..23
// (v4..7); lanes 16-31 hold K=8..15 and K=24..31.
__device__ __forceinline__ v16bf load_frag(const bf16_t* __restrict__ base,
                                           int ld, int lane) {
  const int r   = lane & 15;
  const int sel = lane >> 4;
  const bf16_t* p = base + r * ld + sel * 8;
  v8bf lo = *(const v8bf*)(p);
  v8bf hi = *(const v8bf*)(p + 16);
  v16bf f;
#pragma unroll
  for (int i = 0; i < 8; ++i) { f[i] = lo[i]; f[i + 8] = hi[i]; }
  return f;
}

// Stage one W1 slice (64 h x 32 d) transposed into LDS as [h][d].
// Both global b128 loads are issued before any convert/store.
__device__ __forceinline__ void stage_w1(bf16_t* __restrict__ dst,
                                         const float* __restrict__ W1g,
                                         size_t w1base, int hc, int kk,
                                         int tid) {
  v4f w[2];
#pragma unroll
  for (int i = 0; i < 2; ++i) {
    int idx = i * 256 + tid;   // 512 float4 quads
    int d   = idx >> 4;        // 0..31
    int h4  = (idx & 15) * 4;  // 0..60
    w[i] = *(const v4f*)(W1g + w1base + (size_t)(kk + d) * H_ + hc + h4);
  }
#pragma unroll
  for (int i = 0; i < 2; ++i) {
    int idx = i * 256 + tid;
    int d   = idx >> 4;
    int h4  = (idx & 15) * 4;
#pragma unroll
    for (int j = 0; j < 4; ++j) dst[(h4 + j) * LDW + d] = (bf16_t)w[i][j];
  }
}

__global__ __launch_bounds__(256) void expertnet_fused_kernel(
    const float* __restrict__ zg,  const float* __restrict__ mug,
    const float* __restrict__ W1g, const float* __restrict__ b1g,
    const float* __restrict__ W2g, const float* __restrict__ b2g,
    float* __restrict__ out) {
  extern __shared__ char smem[];
  float*  q_s  = (float*)(smem + SM_Q);    // [TM][K_]
  float*  mu_s = (float*)(smem + SM_MU);   // [K_][NZ_]
  bf16_t* Zs   = (bf16_t*)(smem + SM_Z);   // [TM][LDZ] resident z tile
  bf16_t* W1s0 = (bf16_t*)(smem + SM_W1);  // [HT][LDW] double buffer 0
  bf16_t* W1s1 = W1s0 + HT * LDW;          //            double buffer 1
  bf16_t* W2s  = (bf16_t*)(smem + SM_W2);  // [C_][LDC]  full W2 chunk (B^T)
  bf16_t* h_s  = (bf16_t*)(smem + SM_H);   // [TM][LDC]  q-scaled relu chunk

  const int tid  = threadIdx.x;
  const int lane = tid & 31;
  const int wid  = tid >> 5;  // 8 waves
  const int m0   = blockIdx.x * TM;

  const int sel = lane >> 4;  // C/D layout: M = j + 8*sel
  const int ln  = lane & 15;  // C/D layout: N = ln

  const int wrow  = (wid >> 1) * 32;  // 4 wave-rows of 32
  const int wcol1 = (wid & 1) * 32;   // GEMM1: 2x32 over HT
  const int wcol2 = (wid & 1) * 64;   // GEMM2: 2x64 over C

  // ---- Async DMA of mu (8x512 f32 = 16KB) into LDS, overlapped with z ----
  {
    unsigned lds_base = (unsigned)(uintptr_t)smem;  // low 32b = LDS offset
#pragma unroll
    for (int i = 0; i < 4; ++i) {
      unsigned byteoff = (unsigned)(i * 256 + tid) * 16u;
      unsigned lds_addr = lds_base + SM_MU + byteoff;
      unsigned long long gaddr = (unsigned long long)(uintptr_t)mug + byteoff;
      asm volatile("global_load_async_to_lds_b128 %0, %1, off"
                   :: "v"(lds_addr), "v"(gaddr) : "memory");
    }
  }

  // ---- Stage resident z tile (f32 -> bf16): batch 4 b128 loads, then
  //      4 packed b64 LDS stores, so 4 loads stay in flight. ----
#pragma unroll
  for (int g = 0; g < (TM * NZ_) / (4 * 256) / 4; ++g) {  // 16 groups
    v4f zv[4];
#pragma unroll
    for (int u = 0; u < 4; ++u) {
      int idx = (g * 4 + u) * 256 + tid;
      int r  = idx >> 7;
      int d4 = (idx & 127) * 4;
      zv[u] = *(const v4f*)(zg + (size_t)(m0 + r) * NZ_ + d4);
    }
#pragma unroll
    for (int u = 0; u < 4; ++u) {
      int idx = (g * 4 + u) * 256 + tid;
      int r  = idx >> 7;
      int d4 = (idx & 127) * 4;
      v4bf o;
#pragma unroll
      for (int j = 0; j < 4; ++j) o[j] = (bf16_t)zv[u][j];
      *(v4bf*)(Zs + r * LDZ + d4) = o;
    }
  }
  asm volatile("s_wait_asynccnt 0x0" ::: "memory");  // mu DMA drained
  __syncthreads();

  // ---- Student-t soft assignment q (in-block, deterministic) ----
#pragma unroll
  for (int i = 0; i < (TM * K_) / 256; ++i) {
    int p = i * 256 + tid;
    int r = p >> 3, kc = p & 7;
    float d2 = 0.f;
    for (int d = 0; d < NZ_; ++d) {
      float diff = (float)Zs[r * LDZ + d] - mu_s[kc * NZ_ + d];
      d2 += diff * diff;
    }
    q_s[r * K_ + kc] = 1.0f / (1.0f + d2);
  }
  __syncthreads();
  if (tid < TM) {
    float s = 0.f;
#pragma unroll
    for (int kc = 0; kc < K_; ++kc) s += q_s[tid * K_ + kc];
    float inv = 1.0f / s;
#pragma unroll
    for (int kc = 0; kc < K_; ++kc) q_s[tid * K_ + kc] *= inv;
  }
  __syncthreads();

  // Persistent output accumulators: 2 (row frags) x 4 (col frags) x v8f.
  v8f pacc[2][4];
#pragma unroll
  for (int a = 0; a < 2; ++a)
#pragma unroll
    for (int b = 0; b < 4; ++b)
#pragma unroll
      for (int j = 0; j < 8; ++j) pacc[a][b][j] = 0.f;

  for (int k = 0; k < K_; ++k) {
    const size_t w1base = (size_t)k * NZ_ * H_;
    const size_t w2base = (size_t)k * H_ * C_;

    for (int hc = 0; hc < H_; hc += HT) {
      // -------- GEMM1: h_chunk = z @ W1[:, hc:hc+HT], double-buffered ------
      v8f hacc[2][2];
#pragma unroll
      for (int a = 0; a < 2; ++a)
#pragma unroll
        for (int b = 0; b < 2; ++b)
#pragma unroll
          for (int j = 0; j < 8; ++j) hacc[a][b][j] = 0.f;

      stage_w1(W1s0, W1g, w1base, hc, 0, tid);  // prologue slice kk=0

      for (int kk = 0; kk < NZ_; kk += KT) {
        bf16_t* cur = ((kk >> 5) & 1) ? W1s1 : W1s0;
        bf16_t* nxt = ((kk >> 5) & 1) ? W1s0 : W1s1;
        __syncthreads();  // staged 'cur' visible; prior reads of 'nxt' done
        if (kk + KT < NZ_) {
          stage_w1(nxt, W1g, w1base, hc, kk + KT, tid);
          if (kk + 2 * KT < NZ_) {
            // Prefetch slice kk+64 into L2/WGP$ (global_prefetch_b8).
            int d = lane, seg = wid;  // 32 d-rows x 8 segments of 8 floats
            __builtin_prefetch(
                W1g + w1base + (size_t)(kk + 2 * KT + d) * H_ + hc + seg * 8,
                0, 3);
          }
        }
#pragma unroll
        for (int fm = 0; fm < 2; ++fm) {
          v16bf a  = load_frag(Zs + (wrow + fm * 16) * LDZ + kk, LDZ, lane);
          v16bf b0 = load_frag(cur + (wcol1 + 0)  * LDW, LDW, lane);
          v16bf b1 = load_frag(cur + (wcol1 + 16) * LDW, LDW, lane);
          hacc[fm][0] = WMMA_BF16(a, b0, hacc[fm][0], true);   // next reuses A
          hacc[fm][1] = WMMA_BF16(a, b1, hacc[fm][1], false);
        }
      }

      // Epilogue: bias + ReLU + scale by q[row,k], store bf16 to h_s.
      // (Previous GEMM2's reads of h_s finished before this hc's barriers.)
#pragma unroll
      for (int fm = 0; fm < 2; ++fm) {
#pragma unroll
        for (int fn = 0; fn < 2; ++fn) {
          int hcol = wcol1 + fn * 16 + ln;
          float b1v = b1g[k * H_ + hc + hcol];
#pragma unroll
          for (int j = 0; j < 8; ++j) {
            int mloc = wrow + fm * 16 + j + sel * 8;
            float v = fmaxf(hacc[fm][fn][j] + b1v, 0.f);
            v *= q_s[mloc * K_ + k];
            h_s[mloc * LDC + hcol] = (bf16_t)v;
          }
        }
      }

      // Stage full W2 chunk [64 h][128 c] transposed -> W2s[c][h].
      // Two batches of 4 in-flight b128 loads each.
#pragma unroll
      for (int g = 0; g < 2; ++g) {
        v4f w[4];
#pragma unroll
        for (int u = 0; u < 4; ++u) {
          int idx = (g * 4 + u) * 256 + tid;  // 2048 float4 quads total
          int h2 = idx >> 5;                  // 0..63
          int c4 = (idx & 31) * 4;            // 0..124
          w[u] = *(const v4f*)(W2g + w2base + (size_t)(hc + h2) * C_ + c4);
        }
#pragma unroll
        for (int u = 0; u < 4; ++u) {
          int idx = (g * 4 + u) * 256 + tid;
          int h2 = idx >> 5;
          int c4 = (idx & 31) * 4;
#pragma unroll
          for (int j = 0; j < 4; ++j)
            W2s[(c4 + j) * LDC + h2] = (bf16_t)w[u][j];
        }
      }
      if (hc + HT < H_) {  // prefetch next H-chunk's W2 rows
        __builtin_prefetch(W2g + w2base + (size_t)(hc + HT + wid * 8) * C_ +
                               lane * 4, 0, 3);
      }
      __syncthreads();  // h_s + W2s visible to all waves

      // -------- GEMM2: pacc += h_chunk @ W2[hc:hc+HT, :] -------------------
#pragma unroll
      for (int kk2 = 0; kk2 < HT; kk2 += KT) {
#pragma unroll
        for (int fm = 0; fm < 2; ++fm) {
          v16bf a  = load_frag(h_s + (wrow + fm * 16) * LDC + kk2, LDC, lane);
          v16bf b0 = load_frag(W2s + (wcol2 + 0)  * LDC + kk2, LDC, lane);
          v16bf b1 = load_frag(W2s + (wcol2 + 16) * LDC + kk2, LDC, lane);
          v16bf b2 = load_frag(W2s + (wcol2 + 32) * LDC + kk2, LDC, lane);
          v16bf b3 = load_frag(W2s + (wcol2 + 48) * LDC + kk2, LDC, lane);
          pacc[fm][0] = WMMA_BF16(a, b0, pacc[fm][0], true);
          pacc[fm][1] = WMMA_BF16(a, b1, pacc[fm][1], true);
          pacc[fm][2] = WMMA_BF16(a, b2, pacc[fm][2], true);
          pacc[fm][3] = WMMA_BF16(a, b3, pacc[fm][3], false);
        }
      }
    }
  }

  // ---- Final: add sum_k q[n,k]*b2[k,c], store f32 ----
#pragma unroll
  for (int fm = 0; fm < 2; ++fm) {
#pragma unroll
    for (int j = 0; j < 8; ++j) {
      int mloc = wrow + fm * 16 + j + sel * 8;
      float qrow[K_];
#pragma unroll
      for (int k = 0; k < K_; ++k) qrow[k] = q_s[mloc * K_ + k];
#pragma unroll
      for (int fn = 0; fn < 4; ++fn) {
        int c = wcol2 + fn * 16 + ln;
        float v = pacc[fm][fn][j];
#pragma unroll
        for (int k = 0; k < K_; ++k) v += qrow[k] * b2g[k * C_ + c];
        out[(size_t)(m0 + mloc) * C_ + c] = v;
      }
    }
  }
}

extern "C" void kernel_launch(void* const* d_in, const int* in_sizes, int n_in,
                              void* d_out, int out_size, void* d_ws,
                              size_t ws_size, hipStream_t stream) {
  const float* z  = (const float*)d_in[0];
  const float* mu = (const float*)d_in[1];
  const float* W1 = (const float*)d_in[2];
  const float* b1 = (const float*)d_in[3];
  const float* W2 = (const float*)d_in[4];
  const float* b2 = (const float*)d_in[5];
  float* out = (float*)d_out;

  dim3 grid(N_ / TM);  // 256 blocks
  dim3 block(256);     // 8 waves (wave32)
  expertnet_fused_kernel<<<grid, block, SM_BYTES, stream>>>(z, mu, W1, b1, W2,
                                                            b2, out);
}
